// SingleLSTMCellModel_30116310679823
// MI455X (gfx1250) — matrix operand order repaired and checked by reading, not compiled, not run on hardware
//
#include <hip/hip_runtime.h>

// Problem constants (match reference)
#define SEQ   2048
#define BATCH 64
#define NIN   512
#define NH    512
#define NG    2048   // 4*H gate rows

typedef __attribute__((ext_vector_type(16))) __bf16       v16bf;
typedef __attribute__((ext_vector_type(8)))  float        v8f;
typedef __attribute__((ext_vector_type(4)))  unsigned int v4u;
typedef __attribute__((ext_vector_type(8)))  unsigned int v8u;

// ---------------- workspace layout (bytes) ----------------
static constexpr size_t OFF_WIH   = 0;                                    // NG*NIN bf16 (2 MB)
static constexpr size_t OFF_WHH   = OFF_WIH   + (size_t)NG * NIN * 2;     // NG*NH  bf16 (2 MB)
static constexpr size_t OFF_BIAS  = OFF_WHH   + (size_t)NG * NH * 2;      // NG f32
static constexpr size_t OFF_H0    = OFF_BIAS  + (size_t)NG * 4;           // BATCH*NH bf16
static constexpr size_t OFF_H1    = OFF_H0    + (size_t)BATCH * NH * 2;   // BATCH*NH bf16
static constexpr size_t OFF_CTR   = OFF_H1    + (size_t)BATCH * NH * 2;   // barrier counter
static constexpr size_t OFF_XBF   = OFF_CTR   + 256;                      // SEQ*BATCH*NIN bf16 (128 MB)
static constexpr size_t OFF_XPROJ = OFF_XBF   + (size_t)SEQ * BATCH * NIN * 2; // SEQ*BATCH*NG f32 (1 GB)

// ---------------- fragment helpers (ISA 7.12.2 layouts) ----------------
struct A2frag { v4u lo, hi; };

// 16-bit A 16x32: lane ln holds row m=ln; VGPR0..3 = K [half*8 .. half*8+7],
// VGPR4..7 = K [16+half*8 .. 16+half*8+7]. Pointer p is pre-offset by half*8.
__device__ __forceinline__ v16bf load_frag_a(const unsigned short* p) {
    A2frag t;
    t.lo = *(const v4u*)(p);        // 8 bf16, 16B
    t.hi = *(const v4u*)(p + 16);   // 8 bf16 at K+16
    return __builtin_bit_cast(v16bf, t);
}

// 16-bit B 32x16: lane ln holds col n=ln; VGPR0..3 = K [half*8 .. half*8+7],
// VGPR4..7 = K [16+half*8 .. 16+half*8+7]. Pointer p is pre-offset by half*8.
// Address-space agnostic: used for both global (phase 1) and LDS (phase 2).
__device__ __forceinline__ v16bf load_frag_b(const unsigned short* p) {
    A2frag t;
    t.lo = *(const v4u*)(p);
    t.hi = *(const v4u*)(p + 16);
    return __builtin_bit_cast(v16bf, t);
}

__device__ __forceinline__ v8f wmma_bf16(v16bf a, v16bf b, v8f c) {
    return __builtin_amdgcn_wmma_f32_16x16x32_bf16(
        /*neg_a=*/false, a, /*neg_b=*/false, b,
        /*c_mod=*/(short)0, c, /*reuse_a=*/false, /*reuse_b=*/false);
}

__device__ __forceinline__ unsigned short f32_to_bf16(float x) {
    unsigned u = __float_as_uint(x);
    unsigned r = (u + 0x7FFFu + ((u >> 16) & 1u)) >> 16;  // RNE
    return (unsigned short)r;
}

__device__ __forceinline__ float sigf(float x)  { return 1.0f / (1.0f + __expf(-x)); }
__device__ __forceinline__ float tanhfast(float x) {
    float e = __expf(2.0f * x);
    return 1.0f - 2.0f / (e + 1.0f);   // safe at +/-inf
}

// ---------------- kernels ----------------

// fp32 -> bf16 (grid-stride)
__global__ void k_cvt_bf16(const float* __restrict__ src,
                           unsigned short* __restrict__ dst, long n) {
    long i = (long)blockIdx.x * blockDim.x + threadIdx.x;
    long stride = (long)gridDim.x * blockDim.x;
    for (; i < n; i += stride) dst[i] = f32_to_bf16(src[i]);
}

// bias sum, zero h0, zero barrier counter
__global__ void k_prep(const float* __restrict__ bih, const float* __restrict__ bhh,
                       float* __restrict__ bias, unsigned short* __restrict__ h0,
                       unsigned* __restrict__ ctr) {
    int i = blockIdx.x * blockDim.x + threadIdx.x;
    if (i < NG) bias[i] = bih[i] + bhh[i];
    if (i < BATCH * NH) h0[i] = 0;
    if (i == 0) *ctr = 0u;
}

// Phase 1: x_proj[m][n] = sum_k input_bf[m][k] * Wih_bf[n][k] + bias[n]
// M = SEQ*BATCH = 131072, N = NG = 2048, K = NIN = 512.
// Block = 256 thr (8 waves), block tile 64(M) x 128(N); wave tile 32x32 (2x2 WMMA tiles).
__global__ __launch_bounds__(256) void k_xproj_gemm(
    const unsigned short* __restrict__ Abf,   // (SEQ*BATCH, NIN) bf16 row-major
    const unsigned short* __restrict__ Bbf,   // (NG, NIN) bf16 row-major (W_ih)
    const float* __restrict__ bias,
    float* __restrict__ xproj) {              // (SEQ*BATCH, NG) f32
    const int bx   = blockIdx.x;
    const int nblk = bx & 15;    // 16 N-blocks of 128
    const int mblk = bx >> 4;    // 2048 M-blocks of 64
    const int w    = threadIdx.x >> 5;
    const int lane = threadIdx.x & 31;
    const int ln   = lane & 15;
    const int hf   = lane >> 4;

    const int mbase = mblk * 64  + (w >> 2) * 32;  // wave's 2 M-tiles
    const int nbase = nblk * 128 + (w & 3) * 32;   // wave's 2 N-tiles

    const unsigned short* a0p = Abf + (size_t)(mbase + ln) * NIN + hf * 8;
    const unsigned short* a1p = a0p + (size_t)16 * NIN;
    const unsigned short* b0p = Bbf + (size_t)(nbase + ln) * NIN + hf * 8;
    const unsigned short* b1p = b0p + (size_t)16 * NIN;

    v8f c00 = {0,0,0,0,0,0,0,0}, c01 = {0,0,0,0,0,0,0,0};
    v8f c10 = {0,0,0,0,0,0,0,0}, c11 = {0,0,0,0,0,0,0,0};

#pragma unroll 4
    for (int k0 = 0; k0 < NIN; k0 += 32) {
        v16bf a0 = load_frag_a(a0p + k0);
        v16bf a1 = load_frag_a(a1p + k0);
        v16bf b0 = load_frag_b(b0p + k0);
        v16bf b1 = load_frag_b(b1p + k0);
        c00 = wmma_bf16(a0, b0, c00);
        c01 = wmma_bf16(a0, b1, c01);
        c10 = wmma_bf16(a1, b0, c10);
        c11 = wmma_bf16(a1, b1, c11);
    }

    const int n0 = nbase + ln, n1 = n0 + 16;
    const float bv0 = bias[n0], bv1 = bias[n1];
    float* r0 = xproj + (size_t)(mbase + hf * 8) * NG;  // row m = mbase + v + 8*hf
#pragma unroll
    for (int v = 0; v < 8; ++v) {
        float* rowA = r0 + (size_t)v * NG;
        float* rowB = rowA + (size_t)16 * NG;
        rowA[n0] = c00[v] + bv0;  rowA[n1] = c01[v] + bv1;
        rowB[n0] = c10[v] + bv0;  rowB[n1] = c11[v] + bv1;
    }
}

// Phase 2: persistent recurrent scan. 16 blocks x 256 thr = 128 waves.
// Block owns 2 hidden slices (32 hidden units); its 128 W_hh gate rows
// (128 KB bf16) live in LDS for all 2048 steps. Wave (mt, s_local) computes
// batch tile mt*16.. against hidden slice (blockIdx*2+s_local); cell state c
// stays in registers for the whole sequence.
#define NBLK_SCAN 16
__global__ __launch_bounds__(256) void k_lstm_scan(
    const unsigned short* __restrict__ Whhbf,  // (NG, NH) bf16 row-major
    const float* __restrict__ xproj,           // (SEQ*BATCH, NG) f32 (bias folded)
    unsigned short* __restrict__ h0,           // (BATCH, NH) bf16, holds h_0 = 0
    unsigned short* __restrict__ h1,
    unsigned* __restrict__ ctr,
    float* __restrict__ out) {                 // outputs | h_n | c_n
    // LDS-resident W_hh slice: 128 rows x 512 bf16 = 128 KB (of 320 KB/WGP).
    // LDS row r = s_local*64 + gate*16 + row_in_tile  <->  global row
    //             gate*NH + (blockIdx*2+s_local)*16 + row_in_tile
    __shared__ unsigned short smemW[128 * NH];

    const int wl   = threadIdx.x >> 5;   // wave in block: 0..7
    const int lane = threadIdx.x & 31;
    const int ln   = lane & 15;
    const int hf   = lane >> 4;
    const int mt   = wl & 3;             // batch tile 0..3
    const int sl   = wl >> 2;            // local hidden slice 0..1
    const int hs   = blockIdx.x * 2 + sl;
    const int mbase = mt * 16;
    const int hbase = hs * 16;
    const int ncol  = hbase + ln;

    // ---- cooperative fill of LDS W_hh slice (once; resident all 2048 steps) ----
    {
        v4u* dst = (v4u*)smemW;                       // 8192 x 16B
        for (int u = threadIdx.x; u < 8192; u += 256) {
            const int r  = u >> 6;                    // LDS row 0..127
            const int c  = u & 63;                    // 16B chunk within row
            const int s2 = r >> 6;                    // slice 0/1
            const int g  = (r >> 4) & 3;              // gate
            const int ri = r & 15;
            const int grow = g * NH + (blockIdx.x * 2 + s2) * 16 + ri;
            dst[u] = *((const v4u*)(Whhbf + (size_t)grow * NH) + c);
        }
    }
    __syncthreads();

    // B fragment base pointers into LDS (pre-offset by half*8)
    const unsigned short* Bp0 = smemW + (size_t)(sl * 64 + 0 * 16 + ln) * NH + hf * 8;
    const unsigned short* Bp1 = smemW + (size_t)(sl * 64 + 1 * 16 + ln) * NH + hf * 8;
    const unsigned short* Bp2 = smemW + (size_t)(sl * 64 + 2 * 16 + ln) * NH + hf * 8;
    const unsigned short* Bp3 = smemW + (size_t)(sl * 64 + 3 * 16 + ln) * NH + hf * 8;

    const unsigned short* A0 = h0 + (size_t)(mbase + ln) * NH + hf * 8;
    const unsigned short* A1 = h1 + (size_t)(mbase + ln) * NH + hf * 8;

    v8f cst = {0,0,0,0,0,0,0,0};  // resident cell state (8 rows x this lane's column)

    for (int t = 0; t < SEQ; ++t) {
        const unsigned short* Ap = (t & 1) ? A1 : A0;
        unsigned short* hdst     = (t & 1) ? h0 : h1;

        const float* xp = xproj + ((size_t)t * BATCH + mbase + hf * 8) * NG;
        float* orow     = out   + ((size_t)t * BATCH + mbase + hf * 8) * NH;

        // Prefetch this step's x_proj operands behind the WMMA K-loop.
#pragma unroll
        for (int v = 0; v < 8; ++v) {
            const float* xr = xp + (size_t)v * NG;
            __builtin_prefetch(xr + ncol,          0, 1);
            __builtin_prefetch(xr + NH + ncol,     0, 1);
            __builtin_prefetch(xr + 2 * NH + ncol, 0, 1);
            __builtin_prefetch(xr + 3 * NH + ncol, 0, 1);
        }

        v8f g0 = {0,0,0,0,0,0,0,0}, g1 = {0,0,0,0,0,0,0,0};
        v8f g2 = {0,0,0,0,0,0,0,0}, g3 = {0,0,0,0,0,0,0,0};

#pragma unroll 2
        for (int k0 = 0; k0 < NH; k0 += 32) {
            v16bf a = load_frag_a(Ap + k0);              // h from global (L2)
            g0 = wmma_bf16(a, load_frag_b(Bp0 + k0), g0);// W_hh from LDS (ds_load)
            g1 = wmma_bf16(a, load_frag_b(Bp1 + k0), g1);
            g2 = wmma_bf16(a, load_frag_b(Bp2 + k0), g2);
            g3 = wmma_bf16(a, load_frag_b(Bp3 + k0), g3);
        }

#pragma unroll
        for (int v = 0; v < 8; ++v) {
            const float* xr = xp + (size_t)v * NG;
            float gi = g0[v] + xr[ncol];
            float gf = g1[v] + xr[NH + ncol];
            float gg = g2[v] + xr[2 * NH + ncol];
            float go = g3[v] + xr[3 * NH + ncol];
            float iv = sigf(gi);
            float fv = sigf(gf);
            float gv = tanhfast(gg);
            float ov = sigf(go);
            float cn = fv * cst[v] + iv * gv;
            cst[v] = cn;
            float hv = ov * tanhfast(cn);

            orow[(size_t)v * NH + ncol] = hv;
            const int gm = mbase + hf * 8 + v;
            hdst[(size_t)gm * NH + ncol] = f32_to_bf16(hv);
            if (t == SEQ - 1) {
                out[(size_t)SEQ * BATCH * NH + (size_t)gm * NH + ncol] = hv;                       // h_n
                out[(size_t)SEQ * BATCH * NH + (size_t)BATCH * NH + (size_t)gm * NH + ncol] = cn;  // c_n
            }
        }

        // ---- grid barrier across NBLK_SCAN blocks (release h, acquire peers' h) ----
        __threadfence();
        __syncthreads();
        if (threadIdx.x == 0) {
            atomicAdd(ctr, 1u);
            const unsigned target = (unsigned)(t + 1) * (unsigned)NBLK_SCAN;
            while (atomicAdd(ctr, 0u) < target) __builtin_amdgcn_s_sleep(2);
        }
        __syncthreads();
        __threadfence();
    }
}

// ---------------- host launcher ----------------
extern "C" void kernel_launch(void* const* d_in, const int* in_sizes, int n_in,
                              void* d_out, int out_size, void* d_ws, size_t ws_size,
                              hipStream_t stream) {
    (void)in_sizes; (void)n_in; (void)out_size; (void)ws_size;

    const float* input_seq = (const float*)d_in[0];  // (SEQ, BATCH, NIN)
    const float* W_ih      = (const float*)d_in[1];  // (NG, NIN)
    const float* W_hh      = (const float*)d_in[2];  // (NG, NH)
    const float* b_ih      = (const float*)d_in[3];  // (NG)
    const float* b_hh      = (const float*)d_in[4];  // (NG)
    float* out             = (float*)d_out;

    char* ws = (char*)d_ws;
    unsigned short* wih_bf = (unsigned short*)(ws + OFF_WIH);
    unsigned short* whh_bf = (unsigned short*)(ws + OFF_WHH);
    float*          bias   = (float*)(ws + OFF_BIAS);
    unsigned short* h0     = (unsigned short*)(ws + OFF_H0);
    unsigned short* h1     = (unsigned short*)(ws + OFF_H1);
    unsigned*       ctr    = (unsigned*)(ws + OFF_CTR);
    unsigned short* x_bf   = (unsigned short*)(ws + OFF_XBF);
    float*          xproj  = (float*)(ws + OFF_XPROJ);

    // 1) precision conversion to bf16
    k_cvt_bf16<<<8192, 256, 0, stream>>>(input_seq, x_bf, (long)SEQ * BATCH * NIN);
    k_cvt_bf16<<<1024, 256, 0, stream>>>(W_ih, wih_bf, (long)NG * NIN);
    k_cvt_bf16<<<1024, 256, 0, stream>>>(W_hh, whh_bf, (long)NG * NH);

    // 2) bias sum, h_0 = 0, barrier counter = 0 (must reset every call)
    k_prep<<<(BATCH * NH + 255) / 256, 256, 0, stream>>>(b_ih, b_hh, bias, h0, ctr);

    // 3) big input-projection GEMM: (131072 x 2048 x 512)
    k_xproj_gemm<<<32768, 256, 0, stream>>>(x_bf, wih_bf, bias, xproj);

    // 4) persistent recurrent scan (16 co-resident blocks, LDS-resident W_hh slice,
    //    grid barrier per step)
    k_lstm_scan<<<NBLK_SCAN, 256, 0, stream>>>(whh_bf, xproj, h0, h1, ctr, out);
}
